// CausalSelfAttention_6880537608414
// MI455X (gfx1250) — matrix operand order repaired and checked by reading, not compiled, not run on hardware
//
#include <hip/hip_runtime.h>
#include <hip/hip_bf16.h>

typedef __attribute__((ext_vector_type(16))) __bf16   v16bf;
typedef __attribute__((ext_vector_type(8)))  float    v8f;
typedef __attribute__((ext_vector_type(4)))  unsigned int v4u;

union Frag { v4u u[2]; v16bf b; };

__device__ inline unsigned short f2bf(float f) {
    unsigned u = __builtin_bit_cast(unsigned, f);
    unsigned r = (u + 0x7fffu + ((u >> 16) & 1u)) >> 16;   // RNE
    return (unsigned short)r;
}

__device__ inline v8f vzero8() {
    v8f v;
    #pragma unroll
    for (int i = 0; i < 8; ++i) v[i] = 0.0f;
    return v;
}

__device__ inline v8f wmma_bf16(const Frag& a, const Frag& b, v8f c) {
    return __builtin_amdgcn_wmma_f32_16x16x32_bf16(false, a.b, false, b.b,
                                                   (short)0, c, false, false);
}

// Direct memory->LDS async copy (CDNA5 TDM-adjacent path, ASYNCcnt-tracked).
// LDS aperture truncates flat addr to [31:0], so a 32-bit cast of a __shared__
// pointer is the correct VDST (LDS byte address) operand.
__device__ inline void async_copy_b128(const unsigned short* g, unsigned short* l) {
    unsigned lds = (unsigned)(unsigned long long)l;
    asm volatile("global_load_async_to_lds_b128 %0, %1, off"
                 :: "v"(lds), "v"(g) : "memory");
}
__device__ inline void wait_async0() {
    asm volatile("s_wait_asynccnt 0x0" ::: "memory");
}

// ---------------------------------------------------------------------------
// One-time fp32 -> bf16 elementwise convert (4 elements / thread).
// ---------------------------------------------------------------------------
__global__ __launch_bounds__(256) void convert_kernel(
    const float* __restrict__ src, unsigned short* __restrict__ dst, int n4)
{
    int i = blockIdx.x * 256 + threadIdx.x;
    if (i < n4) {
        float4 v = ((const float4*)src)[i];
        uint2 o;
        o.x = f2bf(v.x) | ((unsigned)f2bf(v.y) << 16);
        o.y = f2bf(v.z) | ((unsigned)f2bf(v.w) << 16);
        ((uint2*)dst)[i] = o;
    }
}

// ---------------------------------------------------------------------------
// One-time W[K=1024][N] fp32 -> Wt[N][1024] bf16 (transpose + convert).
// ---------------------------------------------------------------------------
__global__ __launch_bounds__(256) void transpose_convert_kernel(
    const float* __restrict__ W, unsigned short* __restrict__ Wt, int N)
{
    size_t t = (size_t)blockIdx.x * 256 + threadIdx.x;  // N*1024 threads
    int k = (int)(t & 1023);
    int n = (int)(t >> 10);
    Wt[(size_t)n * 1024 + k] = f2bf(W[(size_t)k * N + n]);
}

// ---------------------------------------------------------------------------
// GEMM: C[8192 x N] = A[8192 x 1024]bf16 * Bt[N x 1024]bf16^T + bias
// MODE 0: scatter Q [BH,T,64], K [BH,T,64], V transposed [BH,64,T] (N=3072)
// MODE 1: fp32 output + bias to d_out (N=1024)
// Block 256 thr = 8 waves (4x2); tile 128(M) x 128(N) x 32(K).
// Staging via global_load_async_to_lds_b128 (no VGPR transit).
// ---------------------------------------------------------------------------
template <int MODE>
__global__ __launch_bounds__(256) void gemm_kernel(
    const unsigned short* __restrict__ A,    // [8192][1024] bf16
    const unsigned short* __restrict__ Bt,   // [N][1024] bf16
    const float* __restrict__ bias,
    unsigned short* __restrict__ out0, unsigned short* __restrict__ out1,
    unsigned short* __restrict__ out2, float* __restrict__ outf, int N)
{
    __shared__ unsigned short sA[128 * 40];  // [m][k], pitch 40 (16B-aligned rows)
    __shared__ unsigned short sB[128 * 40];  // [n][k], pitch 40

    const int K = 1024;
    const int tid  = threadIdx.x;
    const int wave = tid >> 5, lane = tid & 31;
    const int wm = wave & 3, wn = wave >> 2;        // 4x2 wave grid
    const int l15 = lane & 15;
    const int m0 = blockIdx.y * 128, n0 = blockIdx.x * 128;

    v8f acc[2][4];
    #pragma unroll
    for (int mi = 0; mi < 2; ++mi)
        #pragma unroll
        for (int ni = 0; ni < 4; ++ni) acc[mi][ni] = vzero8();

    for (int k0 = 0; k0 < K; k0 += 32) {
        // ---- stage A and Bt tiles (each 128 rows x 32 k): async mem->LDS ----
        #pragma unroll
        for (int i = 0; i < 2; ++i) {
            int idx = tid + i * 256;                 // 512 8-bf16 chunks each
            int r = idx >> 2, c8 = (idx & 3) * 8;
            async_copy_b128(A  + (size_t)(m0 + r) * K + k0 + c8, &sA[r * 40 + c8]);
            async_copy_b128(Bt + (size_t)(n0 + r) * K + k0 + c8, &sB[r * 40 + c8]);
        }
        wait_async0();
        __syncthreads();

        // ---- fragments + 8 WMMA ----
        Frag a[2], b[4];
        const int kbA = (lane < 16) ? 0 : 16;        // A: k 0-15/16-31 per half
        #pragma unroll
        for (int mi = 0; mi < 2; ++mi) {
            const unsigned short* p = &sA[(wm * 32 + mi * 16 + l15) * 40 + kbA];
            a[mi].u[0] = *(const v4u*)p;
            a[mi].u[1] = *(const v4u*)(p + 8);
        }
        const int kbB = (lane < 16) ? 0 : 16;        // B: k 0-15/16-31 per half
        #pragma unroll
        for (int ni = 0; ni < 4; ++ni) {
            const unsigned short* p = &sB[(wn * 64 + ni * 16 + l15) * 40 + kbB];
            b[ni].u[0] = *(const v4u*)p;
            b[ni].u[1] = *(const v4u*)(p + 8);
        }
        #pragma unroll
        for (int mi = 0; mi < 2; ++mi)
            #pragma unroll
            for (int ni = 0; ni < 4; ++ni)
                acc[mi][ni] = wmma_bf16(a[mi], b[ni], acc[mi][ni]);
        __syncthreads();
    }

    // ---- epilogue ----
    const int hi8 = (lane >= 16) ? 8 : 0;
    #pragma unroll
    for (int mi = 0; mi < 2; ++mi) {
        #pragma unroll
        for (int ni = 0; ni < 4; ++ni) {
            int col = n0 + wn * 64 + ni * 16 + l15;
            float bv = bias[col];
            if constexpr (MODE == 0) {
                int part = col >> 10, c = col & 1023;
                int h = c >> 6, d = c & 63;
                #pragma unroll
                for (int r = 0; r < 8; ++r) {
                    int grow = m0 + wm * 32 + mi * 16 + r + hi8;
                    int bb = grow >> 11, t = grow & 2047;
                    unsigned short val = f2bf(acc[mi][ni][r] + bv);
                    if (part == 2) {            // V stored transposed [BH][64][T]
                        out2[((size_t)(bb * 16 + h) * 64 + d) * 2048 + t] = val;
                    } else {                    // Q, K row-major [BH][T][64]
                        size_t off = ((size_t)(bb * 16 + h) * 2048 + t) * 64 + d;
                        (part ? out1 : out0)[off] = val;
                    }
                }
            } else {
                #pragma unroll
                for (int r = 0; r < 8; ++r) {
                    int grow = m0 + wm * 32 + mi * 16 + r + hi8;
                    outf[(size_t)grow * N + col] = acc[mi][ni][r] + bv;
                }
            }
        }
    }
}

// ---------------------------------------------------------------------------
// Flash attention. Grid (T/128, B*H). 8 waves/block = one 128-query slab of
// one head; waves cooperatively stage the SAME K (32x64) and Vt (64x32) tiles
// into LDS per 32-key step (async mem->LDS copies).
// Q [BH][T][64], K [BH][T][64], Vt [BH][64][T] (all bf16). Out: Y bf16 [B*T][1024].
// ---------------------------------------------------------------------------
__global__ __launch_bounds__(256) void attn_kernel(
    const unsigned short* __restrict__ Q, const unsigned short* __restrict__ Kb,
    const unsigned short* __restrict__ Vt, unsigned short* __restrict__ Y)
{
    __shared__ unsigned short sK[32 * 64];   // [key][d]
    __shared__ unsigned short sV[64 * 32];   // [d][key]
    __shared__ unsigned short sP[8 * 512];   // per-wave P[16][32]

    const int tid = threadIdx.x, wave = tid >> 5, lane = tid & 31;
    unsigned short* pbuf = &sP[wave * 512];
    const int l15 = lane & 15;
    const int hi8 = (lane >= 16) ? 8 : 0;

    const int bh  = blockIdx.y;
    const int bq0 = blockIdx.x * 128;
    const int q0  = bq0 + wave * 16;
    const unsigned short* Qh  = Q  + (size_t)bh * 2048 * 64;
    const unsigned short* Kh  = Kb + (size_t)bh * 2048 * 64;
    const unsigned short* Vth = Vt + (size_t)bh * 64 * 2048;

    // Q fragments (A operand, 16x32 each; two cover d=0..63), kept in VGPRs
    Frag aq[2];
    {
        const unsigned short* qrow = Qh + (size_t)(q0 + l15) * 64;
        const int kb = (lane < 16) ? 0 : 16;
        #pragma unroll
        for (int f = 0; f < 2; ++f) {
            const unsigned short* p = qrow + f * 32 + kb;
            aq[f].u[0] = *(const v4u*)p;
            aq[f].u[1] = *(const v4u*)(p + 8);
        }
    }

    v8f o[4];
    #pragma unroll
    for (int nt = 0; nt < 4; ++nt) o[nt] = vzero8();
    float mrow[8], lrow[8];
    #pragma unroll
    for (int r = 0; r < 8; ++r) { mrow[r] = -1e30f; lrow[r] = 0.0f; }
    const float scale = 0.125f;              // 1/sqrt(64)
    const int db = (lane < 16) ? 0 : 16;

    for (int j = 0; j < bq0 + 128; j += 32) {
        // ---- cooperative staging: K tile and Vt tile, async mem->LDS ----
        {
            int kr = tid >> 3, ko = (tid & 7) * 8;   // sK: 32 rows x 64
            async_copy_b128(Kh + (size_t)(j + kr) * 64 + ko, &sK[kr * 64 + ko]);
            int vr = tid >> 2, vo = (tid & 3) * 8;   // sV: 64 rows x 32
            async_copy_b128(Vth + (size_t)vr * 2048 + j + vo, &sV[vr * 32 + vo]);
        }
        wait_async0();
        __syncthreads();

        if (j < q0 + 16) {                   // this wave's causal range
            // ---- S = Q * K^T (two 16-key groups) ----
            v8f s[2]; s[0] = vzero8(); s[1] = vzero8();
            #pragma unroll
            for (int g = 0; g < 2; ++g) {
                const unsigned short* krow = &sK[(g * 16 + l15) * 64];
                Frag bk0, bk1;
                bk0.u[0] = *(const v4u*)(krow + db);
                bk0.u[1] = *(const v4u*)(krow + db + 8);
                bk1.u[0] = *(const v4u*)(krow + 32 + db);
                bk1.u[1] = *(const v4u*)(krow + 32 + db + 8);
                s[g] = wmma_bf16(aq[0], bk0, s[g]);
                s[g] = wmma_bf16(aq[1], bk1, s[g]);
            }

            // ---- causal mask + online softmax ----
            #pragma unroll
            for (int r = 0; r < 8; ++r) {
                int q = q0 + r + hi8;
                float v0 = (j + l15      <= q) ? s[0][r] * scale : -1e30f;
                float v1 = (j + 16 + l15 <= q) ? s[1][r] * scale : -1e30f;
                float mx = fmaxf(v0, v1);
                #pragma unroll
                for (int d = 8; d > 0; d >>= 1) mx = fmaxf(mx, __shfl_xor(mx, d, 32));
                float mnew  = fmaxf(mrow[r], mx);
                float alpha = __expf(mrow[r] - mnew);
                float p0 = __expf(v0 - mnew), p1 = __expf(v1 - mnew);
                float rs = p0 + p1;
                #pragma unroll
                for (int d = 8; d > 0; d >>= 1) rs += __shfl_xor(rs, d, 32);
                lrow[r] = lrow[r] * alpha + rs;
                mrow[r] = mnew;
                #pragma unroll
                for (int nt = 0; nt < 4; ++nt) o[nt][r] *= alpha;
                int m = r + hi8;
                pbuf[m * 32 + l15]      = f2bf(p0);
                pbuf[m * 32 + 16 + l15] = f2bf(p1);
            }
            asm volatile("s_wait_dscnt 0" ::: "memory");  // same-wave LDS RAW fence

            // ---- O += P * V ----
            Frag pa;
            {
                const int kb = (lane < 16) ? 0 : 16;
                const unsigned short* p = pbuf + l15 * 32 + kb;
                pa.u[0] = *(const v4u*)p;
                pa.u[1] = *(const v4u*)(p + 8);
            }
            #pragma unroll
            for (int nt = 0; nt < 4; ++nt) {
                Frag bv;
                const unsigned short* p = &sV[(nt * 16 + l15) * 32 + db];
                bv.u[0] = *(const v4u*)p;
                bv.u[1] = *(const v4u*)(p + 8);
                o[nt] = wmma_bf16(pa, bv, o[nt]);
            }
        }
        __syncthreads();
    }

    // ---- normalize + write Y[b*2048+q][h*64+d] as bf16 ----
    const int bb = bh >> 4, h = bh & 15;
    #pragma unroll
    for (int r = 0; r < 8; ++r) {
        float inv = 1.0f / lrow[r];
        int q = q0 + r + hi8;
        size_t rowoff = (size_t)(bb * 2048 + q) * 1024 + h * 64;
        #pragma unroll
        for (int nt = 0; nt < 4; ++nt)
            Y[rowoff + nt * 16 + l15] = f2bf(o[nt][r] * inv);
    }
}

extern "C" void kernel_launch(void* const* d_in, const int* in_sizes, int n_in,
                              void* d_out, int out_size, void* d_ws, size_t ws_size,
                              hipStream_t stream) {
    const float* x      = (const float*)d_in[0];   // [4,2048,1024]
    const float* W_attn = (const float*)d_in[1];   // [1024,3072]
    const float* b_attn = (const float*)d_in[2];   // [3072]
    const float* W_proj = (const float*)d_in[3];   // [1024,1024]
    const float* b_proj = (const float*)d_in[4];   // [1024]
    float* out = (float*)d_out;                    // [4,2048,1024]

    const size_t NT = (size_t)8192 * 1024;         // 8M elements
    unsigned short* Xbf = (unsigned short*)d_ws;           // 16 MB
    unsigned short* Wat = Xbf + NT;                        //  6 MB (3072x1024)
    unsigned short* Wpt = Wat + (size_t)3072 * 1024;       //  2 MB (1024x1024)
    unsigned short* Qbf = Wpt + (size_t)1024 * 1024;       // 16 MB
    unsigned short* Kbf = Qbf + NT;                        // 16 MB
    unsigned short* Vtb = Kbf + NT;                        // 16 MB
    unsigned short* Ybf = Vtb + NT;                        // 16 MB  (total 88 MB)

    // 0) one-time conversions: x -> bf16; weights -> bf16 transposed [N][K]
    convert_kernel<<<(int)(NT / 4 / 256), 256, 0, stream>>>(x, Xbf, (int)(NT / 4));
    transpose_convert_kernel<<<3072 * 4, 256, 0, stream>>>(W_attn, Wat, 3072);
    transpose_convert_kernel<<<1024 * 4, 256, 0, stream>>>(W_proj, Wpt, 1024);

    // 1) qkv = x @ W_attn + b_attn  -> Q,K [BH,T,64], V transposed [BH,64,T]
    gemm_kernel<0><<<dim3(24, 64), 256, 0, stream>>>(
        Xbf, Wat, b_attn, Qbf, Kbf, Vtb, nullptr, 3072);
    // 2) flash attention -> Y bf16 [B*T, C]
    attn_kernel<<<dim3(16, 64), 256, 0, stream>>>(Qbf, Kbf, Vtb, Ybf);
    // 3) out = Y @ W_proj + b_proj (fp32)
    gemm_kernel<1><<<dim3(8, 64), 256, 0, stream>>>(
        Ybf, Wpt, b_proj, nullptr, nullptr, nullptr, out, 1024);
}
